// MolecularDiffusionGNN_61297773249033
// MI455X (gfx1250) — compile-verified, matrix operand order checked by reading, not run on hardware
//
#include <hip/hip_runtime.h>

#define NN 50000
#define NE 800000
#define NB 256
#define F_AD 39
#define F_BD 10
#define HH 128
#define NHEAD 4
#define HDIM 32
#define NL 3
#define TDIM 128
#define BN_EPS 1e-5f

typedef __attribute__((ext_vector_type(16))) __bf16 v16bf;
typedef __attribute__((ext_vector_type(8)))  float  v8f;

// ---------- helpers ----------
// order-preserving float -> uint map (for atomicMax-based segment max)
__device__ __forceinline__ unsigned fmapu(float f) {
  unsigned u = __builtin_bit_cast(unsigned, f);
  return (u >> 31) ? ~u : (u | 0x80000000u);
}
__device__ __forceinline__ float funmapu(unsigned u) {
  unsigned b = (u >> 31) ? (u & 0x7FFFFFFFu) : ~u;
  return __builtin_bit_cast(float, b);
}
__device__ __forceinline__ float silu_f(float x) {
  return x * __builtin_amdgcn_rcpf(1.f + __expf(-x));
}

// ---------- weight pre-pack: W[128,128] f32 -> bf16 WMMA fragments ----------
// layout: [ktile(4)][ntile(8)][lane(32)][16 bf16]; fragment element j of lane l:
//   col = ntile*16 + (l&15), k = ktile*32 + ((l>>4)*8) + (j<8 ? j : 16 + j-8)
__global__ void pack_w_k(const float* __restrict__ W, __bf16* __restrict__ Wp) {
  int t = blockIdx.x * blockDim.x + threadIdx.x;   // 4*8*32 = 1024 threads
  if (t >= 1024) return;
  int lane = t & 31;
  int frag = t >> 5;                                // kt*8 + nt
  int kt = frag >> 3, nt = frag & 7;
  int col  = nt * 16 + (lane & 15);
  int koff = (lane >> 4) << 3;
  int kb   = kt * 32;
  __bf16* dst = Wp + (size_t)t * 16;
#pragma unroll
  for (int j = 0; j < 8; ++j) {
    dst[j]     = (__bf16)W[(size_t)(kb + koff + j) * HH + col];
    dst[j + 8] = (__bf16)W[(size_t)(kb + 16 + koff + j) * HH + col];
  }
}

// ---------- WMMA GEMM: C[M,128] = act(A[M,128] @ W + bias + G[gidx[row]]) ----------
// one wave per 16x128 output stripe; whole A stripe hoisted into av[4] up-front,
// B fragments batched per k-step into bv[8] so loads overlap the WMMA chain.
template <int ACT, bool HASB, bool HASG>
__global__ __launch_bounds__(256) void gemm_wmma_bf16(
    const float* __restrict__ A, const __bf16* __restrict__ Wp,
    const float* __restrict__ bias, const float* __restrict__ G,
    const int* __restrict__ gidx, float* __restrict__ C, int M)
{
  const int lane = threadIdx.x & 31;
  const int wave = blockIdx.x * (blockDim.x >> 5) + (threadIdx.x >> 5);
  if (wave >= (M >> 4)) return;                    // wave-uniform guard; EXEC stays full
  const int r = lane & 15;
  const int koff = (lane >> 4) << 3;               // 0 or 8
  const float* arow = A + (size_t)(wave * 16 + r) * HH;

  __builtin_prefetch(Wp + (size_t)lane * 16, 0, 3);

  // hoist + convert the entire 16x128 A stripe (4 fragments, 32 VGPRs bf16)
  v16bf av[4];
#pragma unroll
  for (int kt = 0; kt < 4; ++kt) {
    const float* ap = arow + kt * 32 + koff;
#pragma unroll
    for (int j = 0; j < 8; ++j) {
      av[kt][j]     = (__bf16)ap[j];
      av[kt][j + 8] = (__bf16)ap[16 + j];
    }
  }

  v8f acc[8] = {};
#pragma unroll
  for (int kt = 0; kt < 4; ++kt) {
    const __bf16* wbase = Wp + (size_t)kt * 4096 + (size_t)lane * 16;
    v16bf bv[8];                                   // distinct regs -> clause of b128 loads
#pragma unroll
    for (int nt = 0; nt < 8; ++nt) bv[nt] = *(const v16bf*)(wbase + (size_t)nt * 512);
#pragma unroll
    for (int nt = 0; nt < 8; ++nt)
      acc[nt] = __builtin_amdgcn_wmma_f32_16x16x32_bf16(false, av[kt], false, bv[nt],
                                                        (short)0, acc[nt], false, false);
  }

  const int row0 = wave * 16 + ((lane >> 4) << 3); // lanes 16-31 hold M=8..15
  int gb[8];
  if (HASG) {
#pragma unroll
    for (int v = 0; v < 8; ++v) gb[v] = gidx[row0 + v];
  }
#pragma unroll
  for (int nt = 0; nt < 8; ++nt) {
    const int col = nt * 16 + r;
    float bb = HASB ? bias[col] : 0.f;
#pragma unroll
    for (int v = 0; v < 8; ++v) {
      int row = row0 + v;
      float val = acc[nt][v] + bb;
      if (HASG) val += G[(size_t)gb[v] * HH + col];
      if (ACT == 1) val = silu_f(val);
      C[(size_t)row * HH + col] = val;
    }
  }
}

// ---------- small kernels ----------
__global__ void atom_embed_k(const float* __restrict__ x, const float* __restrict__ W,
                             const float* __restrict__ b, float* __restrict__ h) {
  int t = blockIdx.x * blockDim.x + threadIdx.x;
  if (t >= NN * HH) return;
  int n = t >> 7, c = t & 127;
  float s = b[c];
  const float* xr = x + (size_t)n * F_AD;
#pragma unroll
  for (int k = 0; k < F_AD; ++k) s += xr[k] * W[k * HH + c];
  h[t] = s;
}

__global__ void time_embed_k(const int* __restrict__ t, float* __restrict__ te) {
  int i = blockIdx.x * blockDim.x + threadIdx.x;
  if (i >= NB * 64) return;
  int b = i >> 6, j = i & 63;
  float fr = __expf((float)j * (-9.210340371976184f / 63.f));
  float e = (float)t[b] * fr;
  te[b * TDIM + j]      = __sinf(e);
  te[b * TDIM + 64 + j] = __cosf(e);
}

// Mbuf[k][hh] = sum_d We[k, hh*32+d] * ae[hh*32+d]
__global__ void compute_M_k(const float* __restrict__ We, const float* __restrict__ ae,
                            float* __restrict__ Mb) {
  int t = blockIdx.x * blockDim.x + threadIdx.x;
  if (t >= HH * NHEAD) return;
  int k = t >> 2, hh = t & 3;
  float s = 0.f;
#pragma unroll
  for (int d = 0; d < HDIM; ++d) s += We[(size_t)k * HH + hh * HDIM + d] * ae[hh * HDIM + d];
  Mb[k * NHEAD + hh] = s;
}

// P[bd][hh] = bond_W row bd . Mb col hh ;  q[hh] = bond_b . Mb col hh
__global__ void compute_Pq_k(const float* __restrict__ bW, const float* __restrict__ bb,
                             const float* __restrict__ Mb, float* __restrict__ P,
                             float* __restrict__ q) {
  int t = threadIdx.x;
  if (t < F_BD * NHEAD) {
    int bd = t >> 2, hh = t & 3;
    float s = 0.f;
    for (int k = 0; k < HH; ++k) s += bW[bd * HH + k] * Mb[k * NHEAD + hh];
    P[t] = s;
  } else if (t < (F_BD + 1) * NHEAD) {
    int hh = t - F_BD * NHEAD;
    float s = 0.f;
    for (int k = 0; k < HH; ++k) s += bb[k] * Mb[k * NHEAD + hh];
    q[hh] = s;
  }
}

// s_src[n,hh], s_dst[n,hh] from xh
__global__ void attn_scores_k(const float* __restrict__ xh, const float* __restrict__ as,
                              const float* __restrict__ ad, float* __restrict__ ssrc,
                              float* __restrict__ sdst) {
  int t = blockIdx.x * blockDim.x + threadIdx.x;
  if (t >= NN * NHEAD) return;
  int n = t >> 2, hh = t & 3;
  const float* xr = xh + (size_t)n * HH + hh * HDIM;
  float s0 = 0.f, s1 = 0.f;
#pragma unroll
  for (int d = 0; d < HDIM; ++d) { float v = xr[d]; s0 += v * as[hh * HDIM + d]; s1 += v * ad[hh * HDIM + d]; }
  ssrc[t] = s0; sdst[t] = s1;
}

// pass 1: alpha = leaky_relu(s_src[src]+s_dst[dst]+s_edge), segment max via atomicMax
__global__ void edge_pass1_k(const int* __restrict__ src, const int* __restrict__ dst,
                             const float* __restrict__ eattr, const float* __restrict__ P,
                             const float* __restrict__ q, const float* __restrict__ ssrc,
                             const float* __restrict__ sdst, float* __restrict__ alpha,
                             unsigned* __restrict__ amax) {
  int e = blockIdx.x * blockDim.x + threadIdx.x;
  if (e >= NE) return;
  int s = src[e], d = dst[e];
  float ea[F_BD];
#pragma unroll
  for (int i = 0; i < F_BD; ++i) ea[i] = eattr[(size_t)e * F_BD + i];
#pragma unroll
  for (int hh = 0; hh < NHEAD; ++hh) {
    float se = q[hh];
#pragma unroll
    for (int i = 0; i < F_BD; ++i) se += ea[i] * P[i * NHEAD + hh];
    float a = ssrc[s * NHEAD + hh] + sdst[d * NHEAD + hh] + se;
    a = a > 0.f ? a : 0.2f * a;
    alpha[(size_t)e * NHEAD + hh] = a;
    atomicMax(&amax[d * NHEAD + hh], fmapu(a));
  }
}

// pass 2: ea = exp(alpha - amax[dst]); denom[dst] += ea   (alpha overwritten in place)
__global__ void edge_pass2_k(const int* __restrict__ dst, float* alpha,
                             const unsigned* __restrict__ amax, float* __restrict__ denom) {
  int t = blockIdx.x * blockDim.x + threadIdx.x;
  if (t >= NE * NHEAD) return;
  int e = t >> 2, hh = t & 3;
  int d = dst[e];
  float m = funmapu(amax[d * NHEAD + hh]);
  float v = __expf(alpha[t] - m);
  alpha[t] = v;
  atomicAdd(&denom[d * NHEAD + hh], v);
}

// pass 3: agg[dst, c] += xh[src, c] * ea/(denom+eps)
__global__ void edge_scatter_k(const int* __restrict__ src, const int* __restrict__ dst,
                               const float* __restrict__ xh, const float* __restrict__ ea,
                               const float* __restrict__ denom, float* __restrict__ agg) {
  long long t = (long long)blockIdx.x * blockDim.x + threadIdx.x;
  if (t >= (long long)NE * HH) return;
  int e = (int)(t >> 7);
  int c = (int)(t & 127);
  int hh = c >> 5;
  int s = src[e], d = dst[e];
  float w = ea[(size_t)e * NHEAD + hh] * __builtin_amdgcn_rcpf(denom[d * NHEAD + hh] + 1e-16f);
  atomicAdd(&agg[(size_t)d * HH + c], xh[(size_t)s * HH + c] * w);
}

__global__ void bn_stats_k(const float* __restrict__ agg, const float* __restrict__ cb,
                           float* __restrict__ bnsum, float* __restrict__ bnsq) {
  int c = threadIdx.x & 127;
  int sub = threadIdx.x >> 7;               // 0/1
  int stride = gridDim.x * 2;
  float b = cb[c], s = 0.f, s2 = 0.f;
  for (int n = blockIdx.x * 2 + sub; n < NN; n += stride) {
    float v = agg[(size_t)n * HH + c] + b;
    s += v; s2 += v * v;
  }
  atomicAdd(&bnsum[c], s);
  atomicAdd(&bnsq[c], s2);
}

__global__ void bn_final_k(const float* __restrict__ bnsum, const float* __restrict__ bnsq,
                           float* __restrict__ mu, float* __restrict__ rs) {
  int c = threadIdx.x;
  if (c < HH) {
    float m = bnsum[c] * (1.f / NN);
    float v = bnsq[c] * (1.f / NN) - m * m;
    mu[c] = m;
    rs[c] = rsqrtf(v + BN_EPS);
  }
}

// h = relu((agg+cb - mu)*rs*gamma + beta) + h   (residual in place)
__global__ void bn_apply_k(const float* __restrict__ agg, const float* __restrict__ cb,
                           const float* __restrict__ mu, const float* __restrict__ rs,
                           const float* __restrict__ g, const float* __restrict__ bb,
                           float* __restrict__ h) {
  int t = blockIdx.x * blockDim.x + threadIdx.x;
  if (t >= NN * HH) return;
  int c = t & 127;
  float v = agg[t] + cb[c];
  v = (v - mu[c]) * rs[c] * g[c] + bb[c];
  v = v > 0.f ? v : 0.f;
  h[t] = v + h[t];
}

__global__ void count_nodes_k(const int* __restrict__ batch, float* __restrict__ counts) {
  int n = blockIdx.x * blockDim.x + threadIdx.x;
  if (n < NN) atomicAdd(&counts[batch[n]], 1.f);
}

__global__ void pool_sum_k(const float* __restrict__ nf, const int* __restrict__ batch,
                           float* __restrict__ gsum) {
  int t = blockIdx.x * blockDim.x + threadIdx.x;
  if (t >= NN * HH) return;
  int n = t >> 7, c = t & 127;
  atomicAdd(&gsum[(size_t)batch[n] * HH + c], nf[t]);
}

__global__ void pool_div_k(const float* __restrict__ gsum, const float* __restrict__ counts,
                           float* __restrict__ graph) {
  int t = blockIdx.x * blockDim.x + threadIdx.x;
  if (t >= NB * HH) return;
  int b = t >> 7;
  graph[t] = gsum[t] / fmaxf(counts[b], 1.f);
}

__global__ void add_vec_k(const float* __restrict__ a, const float* __restrict__ b,
                          float* __restrict__ c, int n) {
  int t = blockIdx.x * blockDim.x + threadIdx.x;
  if (t < n) c[t] = a[t] + b[t];
}

__global__ void final_out_k(const float* __restrict__ z, const float* __restrict__ W3,
                            const float* __restrict__ b3, float* __restrict__ out) {
  int t = blockIdx.x * blockDim.x + threadIdx.x;
  if (t >= NN * F_AD) return;
  int n = t / F_AD, a = t - n * F_AD;
  float s = b3[a];
  const float* zr = z + (size_t)n * HH;
#pragma unroll 8
  for (int k = 0; k < HH; ++k) s += zr[k] * W3[k * F_AD + a];
  out[t] = s;
}

// ---------- host ----------
static inline int cdiv(long long a, int b) { return (int)((a + b - 1) / b); }

extern "C" void kernel_launch(void* const* d_in, const int* in_sizes, int n_in,
                              void* d_out, int out_size, void* d_ws, size_t ws_size,
                              hipStream_t stream) {
  const float* x       = (const float*)d_in[0];
  const int*   eidx    = (const int*)d_in[1];
  const float* eattr   = (const float*)d_in[2];
  const int*   batch   = (const int*)d_in[3];
  const int*   tarr    = (const int*)d_in[4];
  const float* atom_W  = (const float*)d_in[5];
  const float* atom_b  = (const float*)d_in[6];
  const float* bond_W  = (const float*)d_in[7];
  const float* bond_b  = (const float*)d_in[8];
  const float* lin_W   = (const float*)d_in[9];
  const float* edge_W  = (const float*)d_in[10];
  const float* att_src = (const float*)d_in[11];
  const float* att_dst = (const float*)d_in[12];
  const float* att_edge= (const float*)d_in[13];
  const float* conv_b  = (const float*)d_in[14];
  const float* bn_g    = (const float*)d_in[15];
  const float* bn_b    = (const float*)d_in[16];
  const float* out_W   = (const float*)d_in[17];
  const float* out_b   = (const float*)d_in[18];
  const float* time_W1 = (const float*)d_in[19];
  const float* time_b1 = (const float*)d_in[20];
  const float* time_W2 = (const float*)d_in[21];
  const float* time_b2 = (const float*)d_in[22];
  const float* pool_W1 = (const float*)d_in[23];
  const float* pool_b1 = (const float*)d_in[24];
  const float* pool_W2 = (const float*)d_in[25];
  const float* pool_b2 = (const float*)d_in[26];
  const float* np_W1   = (const float*)d_in[27];
  const float* np_b1   = (const float*)d_in[28];
  const float* np_W2   = (const float*)d_in[29];
  const float* np_b2   = (const float*)d_in[30];
  const float* np_W3   = (const float*)d_in[31];
  const float* np_b3   = (const float*)d_in[32];

  const int* src = eidx;
  const int* dst = eidx + NE;

  float* f = (float*)d_ws;
  size_t off = 0;
  float* hA    = f + off; off += (size_t)NN * HH;        // current node features
  float* xh    = f + off; off += (size_t)NN * HH;        // projection / node_features
  float* agg   = f + off; off += (size_t)NN * HH;        // aggregation / z1 scratch
  float* alpha = f + off; off += (size_t)NE * NHEAD;     // edge attn scratch
  float* ssrc  = f + off; off += (size_t)NN * NHEAD;
  float* sdst  = f + off; off += (size_t)NN * NHEAD;
  float* denom = f + off; off += (size_t)NN * NHEAD;
  unsigned* amax = (unsigned*)(f + off); off += (size_t)NN * NHEAD;
  float* Mbuf  = f + off; off += HH * NHEAD;
  float* Pbuf  = f + off; off += F_BD * NHEAD;
  float* qbuf  = f + off; off += NHEAD;
  float* bnsum = f + off; off += HH;
  float* bnsq  = f + off; off += HH;
  float* bnmu  = f + off; off += HH;
  float* bnrs  = f + off; off += HH;
  float* te    = f + off; off += (size_t)NB * TDIM;
  float* u1    = f + off; off += (size_t)NB * HH;
  float* t2    = f + off; off += (size_t)NB * HH;
  float* counts= f + off; off += NB;
  float* gsum  = f + off; off += (size_t)NB * HH;
  float* graph = f + off; off += (size_t)NB * HH;
  float* p1    = f + off; off += (size_t)NB * HH;
  float* p2    = f + off; off += (size_t)NB * HH;
  float* tg    = f + off; off += (size_t)NB * HH;
  float* Gt    = f + off; off += (size_t)NB * HH;
  off = (off + 7) & ~(size_t)7;                          // 32B align for bf16 fragments
  __bf16* Wp   = (__bf16*)(f + off); off += (size_t)HH * HH / 2;

  auto gb = [](int M) { return ((M >> 4) + 7) / 8; };    // 8 waves / 256-thread block
  #define PACK(Wsrc) pack_w_k<<<4, 256, 0, stream>>>((Wsrc), Wp)

  // time conditioning: te -> silu(te@W1+b1) -> @W2+b2
  time_embed_k<<<cdiv(NB * 64, 256), 256, 0, stream>>>(tarr, te);
  PACK(time_W1);
  gemm_wmma_bf16<1, true, false><<<gb(NB), 256, 0, stream>>>(te, Wp, time_b1, nullptr, nullptr, u1, NB);
  PACK(time_W2);
  gemm_wmma_bf16<0, true, false><<<gb(NB), 256, 0, stream>>>(u1, Wp, time_b2, nullptr, nullptr, t2, NB);

  // atom embedding
  atom_embed_k<<<cdiv((long long)NN * HH, 256), 256, 0, stream>>>(x, atom_W, atom_b, hA);

  for (int l = 0; l < NL; ++l) {
    // xh = h @ lin_W[l]  (WMMA)
    PACK(lin_W + (size_t)l * HH * HH);
    gemm_wmma_bf16<0, false, false><<<gb(NN), 256, 0, stream>>>(hA, Wp, nullptr, nullptr, nullptr, xh, NN);
    attn_scores_k<<<cdiv((long long)NN * NHEAD, 256), 256, 0, stream>>>(
        xh, att_src + l * NHEAD * HDIM, att_dst + l * NHEAD * HDIM, ssrc, sdst);
    // collapsed edge-attention projection (s_edge = edge_attr @ (bond_W @ We @ a_edge))
    compute_M_k<<<2, 256, 0, stream>>>(edge_W + (size_t)l * HH * HH, att_edge + l * NHEAD * HDIM, Mbuf);
    compute_Pq_k<<<1, 64, 0, stream>>>(bond_W, bond_b, Mbuf, Pbuf, qbuf);

    hipMemsetAsync(denom, 0, (size_t)NN * NHEAD * 4, stream);
    hipMemsetAsync(amax, 0, (size_t)NN * NHEAD * 4, stream);   // 0 == mapped "-inf" sentinel
    hipMemsetAsync(agg, 0, (size_t)NN * HH * 4, stream);

    edge_pass1_k<<<cdiv(NE, 256), 256, 0, stream>>>(src, dst, eattr, Pbuf, qbuf, ssrc, sdst, alpha, amax);
    edge_pass2_k<<<cdiv((long long)NE * NHEAD, 256), 256, 0, stream>>>(dst, alpha, amax, denom);
    edge_scatter_k<<<cdiv((long long)NE * HH, 256), 256, 0, stream>>>(src, dst, xh, alpha, denom, agg);

    hipMemsetAsync(bnsum, 0, HH * 4, stream);
    hipMemsetAsync(bnsq, 0, HH * 4, stream);
    bn_stats_k<<<512, 256, 0, stream>>>(agg, conv_b + l * HH, bnsum, bnsq);
    bn_final_k<<<1, 128, 0, stream>>>(bnsum, bnsq, bnmu, bnrs);
    bn_apply_k<<<cdiv((long long)NN * HH, 256), 256, 0, stream>>>(
        agg, conv_b + l * HH, bnmu, bnrs, bn_g + l * HH, bn_b + l * HH, hA);
  }

  // node_features (into xh) = h @ out_W + out_b
  PACK(out_W);
  gemm_wmma_bf16<0, true, false><<<gb(NN), 256, 0, stream>>>(hA, Wp, out_b, nullptr, nullptr, xh, NN);

  // graph mean pool
  hipMemsetAsync(counts, 0, NB * 4, stream);
  hipMemsetAsync(gsum, 0, (size_t)NB * HH * 4, stream);
  count_nodes_k<<<cdiv(NN, 256), 256, 0, stream>>>(batch, counts);
  pool_sum_k<<<cdiv((long long)NN * HH, 256), 256, 0, stream>>>(xh, batch, gsum);
  pool_div_k<<<cdiv(NB * HH, 256), 256, 0, stream>>>(gsum, counts, graph);

  // pool MLP
  PACK(pool_W1);
  gemm_wmma_bf16<1, true, false><<<gb(NB), 256, 0, stream>>>(graph, Wp, pool_b1, nullptr, nullptr, p1, NB);
  PACK(pool_W2);
  gemm_wmma_bf16<0, true, false><<<gb(NB), 256, 0, stream>>>(p1, Wp, pool_b2, nullptr, nullptr, p2, NB);

  // tg = t_emb + graph_emb ; Gt = tg @ np_W1[H:2H] + np_b1  (split-K of the concat GEMM)
  add_vec_k<<<cdiv(NB * HH, 256), 256, 0, stream>>>(t2, p2, tg, NB * HH);
  PACK(np_W1 + (size_t)HH * HH);
  gemm_wmma_bf16<0, true, false><<<gb(NB), 256, 0, stream>>>(tg, Wp, np_b1, nullptr, nullptr, Gt, NB);

  // z1 = silu(nf @ np_W1[:H] + Gt[batch]) -> agg ; z2 = silu(z1 @ np_W2 + b2) -> hA
  PACK(np_W1);
  gemm_wmma_bf16<1, false, true><<<gb(NN), 256, 0, stream>>>(xh, Wp, nullptr, Gt, batch, agg, NN);
  PACK(np_W2);
  gemm_wmma_bf16<1, true, false><<<gb(NN), 256, 0, stream>>>(agg, Wp, np_b2, nullptr, nullptr, hA, NN);

  // out = z2 @ np_W3 + np_b3  (Nc=39 -> scalar epilogue kernel)
  final_out_k<<<cdiv((long long)NN * F_AD, 256), 256, 0, stream>>>(hA, np_W3, np_b3, (float*)d_out);
  #undef PACK
}
